// CaiGPTQLinearOp_62294205661467
// MI455X (gfx1250) — matrix operand order repaired and checked by reading, not compiled, not run on hardware
//
#include <hip/hip_runtime.h>
#include <stdint.h>
#include <stddef.h>

// ---------------------------------------------------------------------------
// Fused GPTQ(4-bit, group=128) dequant + GEMM + bias + residual for gfx1250.
// C[m,n] = sum_k x[m,k] * (q[k,n]-zero[g,n])*scale[g,n] + bias[n] + res[m,n]
// M = B*S = 8192, K = 4096, N = 4096  (derived from in_sizes at launch).
//
// Tile: 128x128 per workgroup, K-step 32 (one v_wmma_f32_16x16x32_bf16).
// 8 waves (wave32); each wave computes a 64x32 sub-tile = 4x2 WMMA tiles.
// Double-buffered LDS staging; x is converted fp32->bf16 via v_cvt_pk,
// weights are dequantized nibble->bf16 (one FMA per element) into LDS.
// ---------------------------------------------------------------------------

typedef __bf16 bf16_t;
typedef bf16_t bf16x2 __attribute__((ext_vector_type(2)));
typedef bf16_t bf16x8 __attribute__((ext_vector_type(8), aligned(16)));
typedef bf16_t v16bf  __attribute__((ext_vector_type(16)));
typedef float  v8f    __attribute__((ext_vector_type(8)));

#define BM 128
#define BN 128
#define BK 32
#define THREADS 256
#define XSTRIDE 40   // 32 bf16 + 8 pad  (80B rows -> conflict-free b128 reads)
#define WSTRIDE 40
#define GROUP_SIZE 128

// two fp32 -> packed bf16 (one v_cvt_pk_bf16_f32 when available)
__device__ __forceinline__ uint32_t cvt_pk_bf16(float lo, float hi) {
#if __has_builtin(__builtin_amdgcn_cvt_pk_bf16_f32)
    bf16x2 r = __builtin_amdgcn_cvt_pk_bf16_f32(lo, hi);
    return __builtin_bit_cast(uint32_t, r);
#else
    bf16x2 r;
    r.x = (bf16_t)lo;
    r.y = (bf16_t)hi;
    return __builtin_bit_cast(uint32_t, r);
#endif
}

__global__ __launch_bounds__(THREADS)
void gptq_wmma_bf16_kernel(const float*   __restrict__ x,
                           const int32_t* __restrict__ wq,
                           const float*   __restrict__ scales,
                           const int32_t* __restrict__ zeros,
                           const float*   __restrict__ bias,
                           const float*   __restrict__ residual,
                           float*         __restrict__ out,
                           int M, int N, int K)
{
    __shared__ uint16_t xs[2][BM * XSTRIDE];   // activations, [m][k] bf16 bits
    __shared__ uint16_t wsm[2][BN * WSTRIDE];  // weights,     [n][k] bf16 bits

    const int tid   = threadIdx.x;
    const int lane  = tid & 31;
    const int wave  = tid >> 5;
    const int mBase = blockIdx.y * BM;
    const int nBase = blockIdx.x * BN;

    // wave sub-tile offsets within the 128x128 block
    const int wm = (wave & 1) * 64;   // 2 waves along M
    const int wn = (wave >> 1) * 32;  // 4 waves along N

    // x loader: two threads per row, 16 floats each
    const int xr = tid >> 1;          // row 0..127
    const int xh = (tid & 1) * 16;    // column half (in elements)

    // weight loader: 512 packed dwords per tile, 2 per thread (same column n)
    const int wkp = tid >> 7;         // packed-k row within tile: 0..1 (+2 for 2nd)
    const int wnn = tid & 127;        // column within tile

    const int KT = K / BK;

    const v8f vzero = {0.f, 0.f, 0.f, 0.f, 0.f, 0.f, 0.f, 0.f};
    v8f acc[4][2];
#pragma unroll
    for (int i = 0; i < 4; ++i)
#pragma unroll
        for (int j = 0; j < 2; ++j)
            acc[i][j] = vzero;

    float4  xstg[4];
    int32_t wstg0, wstg1;
    float   wscale;
    int32_t wzero;

    auto load_global = [&](int kt) {
        const int k0 = kt * BK;
        const float* xp = x + (size_t)(mBase + xr) * K + k0 + xh;
#pragma unroll
        for (int i = 0; i < 4; ++i)
            xstg[i] = ((const float4*)xp)[i];
        const int kp0 = (k0 >> 3) + wkp;   // row in packed weight [K/8, N]
        wstg0  = wq[(size_t)kp0 * N + nBase + wnn];
        wstg1  = wq[(size_t)(kp0 + 2) * N + nBase + wnn];
        const int g = k0 / GROUP_SIZE;     // BK=32 divides GROUP_SIZE=128
        wscale = scales[(size_t)g * N + nBase + wnn];
        wzero  = zeros [(size_t)g * N + nBase + wnn];
    };

    auto store_lds = [&](int buf) {
        // activations: 16 fp32 -> 16 bf16 (8x v_cvt_pk_bf16_f32, 32B store)
        uint32_t px[8];
#pragma unroll
        for (int i = 0; i < 4; ++i) {
            px[2 * i]     = cvt_pk_bf16(xstg[i].x, xstg[i].y);
            px[2 * i + 1] = cvt_pk_bf16(xstg[i].z, xstg[i].w);
        }
        uint32_t* xd = (uint32_t*)&xs[buf][xr * XSTRIDE + xh];
#pragma unroll
        for (int i = 0; i < 8; ++i) xd[i] = px[i];

        // weights: dequant 2 packed dwords -> 2x (8 bf16), K-major contiguous.
        // (q - z) * s  ==  q * s + (-z * s)   -> one FMA per nibble
        const float s   = wscale;
        const float mzs = -(float)wzero * wscale;
#pragma unroll
        for (int half = 0; half < 2; ++half) {
            const uint32_t w = (uint32_t)(half ? wstg1 : wstg0);
            uint32_t pw[4];
#pragma unroll
            for (int j = 0; j < 4; ++j) {
                const float lo = __builtin_fmaf(
                    (float)((w >> (8 * j))     & 15u), s, mzs);
                const float hi = __builtin_fmaf(
                    (float)((w >> (8 * j + 4)) & 15u), s, mzs);
                pw[j] = cvt_pk_bf16(lo, hi);   // K = kbase+2j, kbase+2j+1
            }
            uint32_t* wd =
                (uint32_t*)&wsm[buf][wnn * WSTRIDE + (wkp + 2 * half) * 8];
#pragma unroll
            for (int j = 0; j < 4; ++j) wd[j] = pw[j];
        }
    };

    const int l15   = lane & 15;
    const int akoff = (lane < 16) ? 0 : 8;   // A frag: K chunk base (elements)
    const int bkoff = (lane < 16) ? 0 : 8;   // B frag: K chunk base (elements)

    auto compute = [&](int buf) {
        v16bf bfrag[2];
#pragma unroll
        for (int ni = 0; ni < 2; ++ni) {
            const uint16_t* wr = &wsm[buf][(wn + ni * 16 + l15) * WSTRIDE + bkoff];
            bf16x8 lo = *(const bf16x8*)wr;
            bf16x8 hi = *(const bf16x8*)(wr + 16);
            bfrag[ni] = __builtin_shufflevector(lo, hi, 0, 1, 2, 3, 4, 5, 6, 7,
                                                8, 9, 10, 11, 12, 13, 14, 15);
        }
#pragma unroll
        for (int mi = 0; mi < 4; ++mi) {
            const uint16_t* xr0 = &xs[buf][(wm + mi * 16 + l15) * XSTRIDE];
            bf16x8 lo = *(const bf16x8*)(xr0 + akoff);        // K 0-7 / 8-15
            bf16x8 hi = *(const bf16x8*)(xr0 + akoff + 16);   // K 16-23 / 24-31
            v16bf afrag = __builtin_shufflevector(lo, hi, 0, 1, 2, 3, 4, 5, 6, 7,
                                                  8, 9, 10, 11, 12, 13, 14, 15);
#pragma unroll
            for (int ni = 0; ni < 2; ++ni) {
                acc[mi][ni] = __builtin_amdgcn_wmma_f32_16x16x32_bf16(
                    false, afrag, false, bfrag[ni], (short)0, acc[mi][ni],
                    false, false);
            }
        }
    };

    // ---- software-pipelined main loop (double-buffered LDS) ----
    load_global(0);
    store_lds(0);
    __syncthreads();

    for (int kt = 0; kt < KT; ++kt) {
        const int cur = kt & 1;
        const bool more = (kt + 1 < KT);
        if (more) load_global(kt + 1);   // issue global loads first
        compute(cur);                    // WMMAs hide the load latency
        if (more) store_lds(cur ^ 1);
        __syncthreads();
    }

    // ---- epilogue: + bias + residual, f32 store ----
    const int mo = (lane >> 4) * 8;      // upper lane half -> rows +8
#pragma unroll
    for (int mi = 0; mi < 4; ++mi) {
#pragma unroll
        for (int ni = 0; ni < 2; ++ni) {
            const int ncol = nBase + wn + ni * 16 + l15;
            const float bv = bias[ncol];
            const int mrow = mBase + wm + mi * 16 + mo;
#pragma unroll
            for (int i = 0; i < 8; ++i) {
                const size_t off = (size_t)(mrow + i) * N + ncol;
                out[off] = acc[mi][ni][i] + bv + residual[off];
            }
        }
    }
}

extern "C" void kernel_launch(void* const* d_in, const int* in_sizes, int n_in,
                              void* d_out, int out_size, void* d_ws, size_t ws_size,
                              hipStream_t stream) {
    const float*   x        = (const float*)d_in[0];
    const int32_t* weight   = (const int32_t*)d_in[1];
    const float*   wscales  = (const float*)d_in[2];
    const int32_t* wzeros   = (const int32_t*)d_in[3];
    const float*   bias     = (const float*)d_in[4];
    const float*   residual = (const float*)d_in[5];
    float*         out      = (float*)d_out;

    const int N = in_sizes[4];                        // bias is [N]
    const int K = (int)(((long)in_sizes[1] * 8) / N); // weight is [K/8, N]
    const int M = (int)((long)in_sizes[0] / K);       // x is [B*S, K]

    dim3 grid(N / BN, M / BM);
    gptq_wmma_bf16_kernel<<<grid, THREADS, 0, stream>>>(
        x, weight, wscales, wzeros, bias, residual, out, M, N, K);
}
